// NequipBlock_2259152798306
// MI455X (gfx1250) — matrix operand order repaired and checked by reading, not compiled, not run on hardware
//
#include <hip/hip_runtime.h>
#include <hip/hip_bf16.h>
#include <math.h>

// ---------------------------------------------------------------------------
// Types for CDNA5 WMMA (wave32): v_wmma_f32_16x16x32_bf16
// ---------------------------------------------------------------------------
typedef __bf16 bf16;
typedef __attribute__((ext_vector_type(8)))  __bf16 v8bf;
typedef __attribute__((ext_vector_type(16))) __bf16 v16bf;
typedef __attribute__((ext_vector_type(8)))  float  v8f;

#define MULC 32
#define NBC  8
#define HC   64
#define NWC  128
#define FRAGS_PER_LAYER 28   // w0: 4 (kt=1,nt=4), w1: 8 (kt=2,nt=4), w2: 16 (kt=2,nt=8)
#define FRAG_HALVES 512      // 32 lanes x 16 bf16

// Fast silu: v_exp_f32 + v_rcp_f32 + v_mul (avoids IEEE div_scale/div_fixup chain)
__device__ __forceinline__ float silu(float x) {
    return x * __builtin_amdgcn_rcpf(1.0f + __expf(-x));
}

// A-fragment (16x32 bf16, M=edges) loaded from a row-major LDS tile.
// Per ISA 7.12.2: lanes 0-15 -> M=lane, K={0..7,16..23}; lanes 16-31 -> M=lane-16, K={8..15,24..31}
__device__ __forceinline__ v16bf load_a_frag(const bf16* tile, int ktot, int kk, int lane) {
    int m  = lane & 15;
    int kb = kk * 32 + ((lane < 16) ? 0 : 8);
    const bf16* p = tile + m * ktot + kb;
    v8bf c0 = *(const v8bf*)(p);
    v8bf c1 = *(const v8bf*)(p + 16);
    v16bf a;
#pragma unroll
    for (int i = 0; i < 8; ++i) { a[i] = c0[i]; a[i + 8] = c1[i]; }
    return a;
}

// B-fragment: pre-packed in global memory as [frag][lane][16 halves]
__device__ __forceinline__ v16bf load_b_frag(const bf16* pw, int frag, int lane) {
    return *(const v16bf*)(pw + (size_t)frag * FRAG_HALVES + lane * 16);
}

// Store C-fragment (16x16 f32) with silu into a row-major bf16 LDS tile.
// C layout: lanes 0-15 -> N=lane, VGPR v -> M=v; lanes 16-31 -> N=lane-16, M=8+v
__device__ __forceinline__ void store_c_silu(bf16* tile, int stride, int nn, int lane, v8f c) {
    int col   = nn * 16 + (lane & 15);
    int rbase = (lane < 16) ? 0 : 8;
#pragma unroll
    for (int v = 0; v < 8; ++v) {
        float x = c[v];
        tile[(rbase + v) * stride + col] = (bf16)silu(x);
    }
}

__device__ __forceinline__ void store_c_f32(float* tile, int stride, int nn, int lane, v8f c) {
    int col   = nn * 16 + (lane & 15);
    int rbase = (lane < 16) ? 0 : 8;
#pragma unroll
    for (int v = 0; v < 8; ++v) {
        tile[(rbase + v) * stride + col] = c[v];
    }
}

// ---------------------------------------------------------------------------
// Weight pre-pack: MLP weights -> bf16 B-fragments (scales folded in)
// grid = 2*FRAGS_PER_LAYER blocks, 32 threads
// ---------------------------------------------------------------------------
__global__ void k_pack(const float* __restrict__ w0, const float* __restrict__ w1,
                       const float* __restrict__ w2, bf16* __restrict__ out) {
    int frag = blockIdx.x;           // 0..55
    int lane = threadIdx.x;          // 0..31
    int l = frag / FRAGS_PER_LAYER;
    int f = frag % FRAGS_PER_LAYER;
    const float* W; int Kreal, Nout, kk, nn; float scale;
    if (f < 4) {                      // mlp_w0 [8,64], scale 1/sqrt(8)
        W = w0 + (size_t)l * NBC * HC; Kreal = NBC; Nout = HC; nn = f; kk = 0;
        scale = 0.35355339059327373f;
    } else if (f < 12) {              // mlp_w1 [64,64], scale 1/sqrt(64)
        int g = f - 4;
        W = w1 + (size_t)l * HC * HC;  Kreal = HC; Nout = HC; nn = g >> 1; kk = g & 1;
        scale = 0.125f;
    } else {                          // mlp_w2 [64,128], unscaled
        int g = f - 12;
        W = w2 + (size_t)l * HC * NWC; Kreal = HC; Nout = NWC; nn = g >> 1; kk = g & 1;
        scale = 1.0f;
    }
    int col   = nn * 16 + (lane & 15);
    int kbase = kk * 32 + ((lane < 16) ? 0 : 8);
    bf16* dst = out + (size_t)frag * FRAG_HALVES + lane * 16;
#pragma unroll
    for (int j = 0; j < 16; ++j) {
        int K = kbase + ((j < 8) ? j : j + 8);
        float v = (K < Kreal) ? W[(size_t)K * Nout + col] * scale : 0.0f;
        dst[j] = (bf16)v;
    }
}

// ---------------------------------------------------------------------------
// Node embedding: h_s = onehot @ embed_w / sqrt(S); h_v = 0
// ---------------------------------------------------------------------------
__global__ void k_embed(const int* __restrict__ species, const float* __restrict__ embed_w,
                        float* __restrict__ hs, float* __restrict__ hv, int Nn) {
    int idx = blockIdx.x * blockDim.x + threadIdx.x;
    if (idx >= Nn * MULC) return;
    int n = idx >> 5, u = idx & 31;
    hs[idx] = embed_w[species[n] * MULC + u] * 0.4472135954999579f; // 1/sqrt(5)
    hv[(size_t)idx * 3 + 0] = 0.0f;
    hv[(size_t)idx * 3 + 1] = 0.0f;
    hv[(size_t)idx * 3 + 2] = 0.0f;
}

// ---------------------------------------------------------------------------
// Per-layer node prologue: self-interaction Linear -> gather tables; zero accumulators
// ---------------------------------------------------------------------------
__global__ void k_node_pre(const float* __restrict__ hs_in, const float* __restrict__ hv_in,
                           const float* __restrict__ wl1s, const float* __restrict__ wl1v,
                           float* __restrict__ hsg, float* __restrict__ hvg,
                           float* __restrict__ n0, float* __restrict__ n1, int Nn) {
    int idx = blockIdx.x * blockDim.x + threadIdx.x;
    if (idx >= Nn * MULC) return;
    int n = idx >> 5, u = idx & 31;
    const float inv_sqrt_mul = 0.17677669529663687f; // 1/sqrt(32)
    float acc = 0.0f, a0 = 0.0f, a1 = 0.0f, a2 = 0.0f;
#pragma unroll 8
    for (int v = 0; v < MULC; ++v) {
        float ws = wl1s[v * MULC + u];
        float wv = wl1v[v * MULC + u];
        acc += hs_in[n * MULC + v] * ws;
        const float* hvp = hv_in + ((size_t)n * MULC + v) * 3;
        a0 += hvp[0] * wv; a1 += hvp[1] * wv; a2 += hvp[2] * wv;
    }
    hsg[idx] = acc * inv_sqrt_mul;
    float* hvo = hvg + (size_t)idx * 3;
    hvo[0] = a0 * inv_sqrt_mul; hvo[1] = a1 * inv_sqrt_mul; hvo[2] = a2 * inv_sqrt_mul;
    // zero scatter accumulators
    n0[(size_t)n * 64 + u] = 0.0f;
    n0[(size_t)n * 64 + 32 + u] = 0.0f;
#pragma unroll
    for (int i = 0; i < 3; ++i) {
        n1[(size_t)n * 192 + u * 3 + i] = 0.0f;
        n1[(size_t)n * 192 + 96 + u * 3 + i] = 0.0f;
    }
}

// ---------------------------------------------------------------------------
// Fused edge kernel: bessel/SH -> 3x WMMA MLP -> CG tensor product -> atomic scatter
// One wave per 16-edge tile; 4 waves per block. E must be a multiple of 64.
// ---------------------------------------------------------------------------
#define WPB 4
__global__ __launch_bounds__(128) void k_edge(const float* __restrict__ evec,
                                              const int* __restrict__ senders,
                                              const int* __restrict__ receivers,
                                              const bf16* __restrict__ pw,
                                              const float* __restrict__ hsg,
                                              const float* __restrict__ hvg,
                                              float* __restrict__ n0,
                                              float* __restrict__ n1) {
    __shared__ bf16  actA[WPB][16 * 64];
    __shared__ bf16  actB[WPB][16 * 64];
    __shared__ float wt[WPB][16 * 128];
    __shared__ float sht[WPB][16 * 4];

    int wave = threadIdx.x >> 5;
    int lane = threadIdx.x & 31;
    int tile = blockIdx.x * WPB + wave;
    int e0   = tile * 16;

    bf16* a0t = actA[wave]; // layer-0 input, stride 32 (K padded 8->32)

    // ---- Stage 1: per-edge radial basis + spherical harmonics (lanes 0..15) ----
    if (lane < 16) {
        int e = e0 + lane;
        float x = evec[(size_t)e * 3 + 0];
        float y = evec[(size_t)e * 3 + 1];
        float z = evec[(size_t)e * 3 + 2];
        float r = sqrtf(x * x + y * y + z * z + 1e-12f);
        float inv = 1.0f / r;
        const float sqrt3 = 1.7320508075688772f;
        sht[wave][lane * 4 + 0] = sqrt3 * x * inv;
        sht[wave][lane * 4 + 1] = sqrt3 * y * inv;
        sht[wave][lane * 4 + 2] = sqrt3 * z * inv;
        float xr = r * 0.2f; // r / R_MAX
        float x2 = xr * xr, x3 = x2 * xr, x6 = x3 * x3, x7 = x6 * xr, x8 = x7 * xr;
        float env = (xr < 1.0f) ? (1.0f - 28.0f * x6 + 48.0f * x7 - 21.0f * x8) : 0.0f;
        float pref = 0.6324555320336759f * inv * env; // sqrt(2/5)/r * u(x)
        const float w0f = 0.6283185307179586f;        // pi / 5
#pragma unroll
        for (int nb = 1; nb <= NBC; ++nb) {
            a0t[lane * 32 + (nb - 1)] = (bf16)(pref * __sinf((float)nb * w0f * r));
        }
#pragma unroll
        for (int k = NBC; k < 32; ++k) a0t[lane * 32 + k] = (bf16)0.0f;
    }
    __syncthreads();

    // ---- Stage 2: MLP layer 0  [16x32] x [32x64] ----
    {
        v16bf a = load_a_frag(a0t, 32, 0, lane);
#pragma unroll
        for (int nn = 0; nn < 4; ++nn) {
            v16bf b = load_b_frag(pw, nn, lane);
            v8f c;
#pragma unroll
            for (int i = 0; i < 8; ++i) c[i] = 0.0f;
            c = __builtin_amdgcn_wmma_f32_16x16x32_bf16(false, a, false, b,
                                                        (short)0, c, false, false);
            store_c_silu(actB[wave], 64, nn, lane, c);
        }
    }
    __syncthreads();

    // ---- Stage 3: MLP layer 1  [16x64] x [64x64] ----
#pragma unroll
    for (int nn = 0; nn < 4; ++nn) {
        v8f c;
#pragma unroll
        for (int i = 0; i < 8; ++i) c[i] = 0.0f;
#pragma unroll
        for (int kk = 0; kk < 2; ++kk) {
            v16bf a = load_a_frag(actB[wave], 64, kk, lane);
            v16bf b = load_b_frag(pw, 4 + nn * 2 + kk, lane);
            c = __builtin_amdgcn_wmma_f32_16x16x32_bf16(false, a, false, b,
                                                        (short)0, c, false, false);
        }
        store_c_silu(actA[wave], 64, nn, lane, c);
    }
    __syncthreads();

    // ---- Stage 4: MLP layer 2  [16x64] x [64x128] -> per-edge TP weights ----
#pragma unroll
    for (int nn = 0; nn < 8; ++nn) {
        v8f c;
#pragma unroll
        for (int i = 0; i < 8; ++i) c[i] = 0.0f;
#pragma unroll
        for (int kk = 0; kk < 2; ++kk) {
            v16bf a = load_a_frag(actA[wave], 64, kk, lane);
            v16bf b = load_b_frag(pw, 12 + nn * 2 + kk, lane);
            c = __builtin_amdgcn_wmma_f32_16x16x32_bf16(false, a, false, b,
                                                        (short)0, c, false, false);
        }
        store_c_f32(wt[wave], 128, nn, lane, c);
    }
    __syncthreads();

    // ---- Stage 5: Clebsch-Gordan uvu TP + scatter (lane = channel u) ----
    {
        const int u = lane;
        const float inv_sqrt3 = 0.5773502691896258f;
        const float* wrow = wt[wave];
        for (int t = 0; t < 16; ++t) {
            int e   = e0 + t;
            int s   = senders[e];
            int rcv = receivers[e];
            float w1 = wrow[t * 128 + u];
            float w2 = wrow[t * 128 + 32 + u];
            float w3 = wrow[t * 128 + 64 + u];
            float w4 = wrow[t * 128 + 96 + u];
            float es = hsg[(size_t)s * MULC + u];
            const float* evp = hvg + ((size_t)s * MULC + u) * 3;
            float ev0 = evp[0], ev1 = evp[1], ev2 = evp[2];
            float s0 = sht[wave][t * 4 + 0];
            float s1 = sht[wave][t * 4 + 1];
            float s2 = sht[wave][t * 4 + 2];
            float o0a  = w1 * es;
            float o0b  = w2 * (ev0 * s0 + ev1 * s1 + ev2 * s2) * inv_sqrt3;
            float w3es = w3 * es;
            atomicAdd(&n0[(size_t)rcv * 64 + u],       o0a);
            atomicAdd(&n0[(size_t)rcv * 64 + 32 + u],  o0b);
            atomicAdd(&n1[(size_t)rcv * 192 + u * 3 + 0],      w3es * s0);
            atomicAdd(&n1[(size_t)rcv * 192 + u * 3 + 1],      w3es * s1);
            atomicAdd(&n1[(size_t)rcv * 192 + u * 3 + 2],      w3es * s2);
            atomicAdd(&n1[(size_t)rcv * 192 + 96 + u * 3 + 0], w4 * ev0);
            atomicAdd(&n1[(size_t)rcv * 192 + 96 + u * 3 + 1], w4 * ev1);
            atomicAdd(&n1[(size_t)rcv * 192 + 96 + u * 3 + 2], w4 * ev2);
        }
    }
}

// ---------------------------------------------------------------------------
// Per-layer node epilogue: lin2 + residual FCTP + gated nonlinearity
// ---------------------------------------------------------------------------
__global__ void k_node_post(const float* __restrict__ hs_in, const float* __restrict__ hv_in,
                            const float* __restrict__ n0, const float* __restrict__ n1,
                            const int* __restrict__ species,
                            const float* __restrict__ wl2s, const float* __restrict__ wl2v,
                            const float* __restrict__ wrs, const float* __restrict__ wrv,
                            float* __restrict__ hs_out, float* __restrict__ hv_out, int Nn) {
    int idx = blockIdx.x * blockDim.x + threadIdx.x;
    if (idx >= Nn * MULC) return;
    int n = idx >> 5, u = idx & 31;
    int sp = species[n];
    const float c1 = 1.0f / (16.0f * 8.0f);     // 1/AVG_NEIGH / sqrt(2*MUL)
    const float c2 = 0.07905694150420949f;      // 1/sqrt(MUL*S) = 1/sqrt(160)

    float sf = 0.0f, sg = 0.0f;
    float v0 = 0.0f, v1 = 0.0f, v2 = 0.0f;
#pragma unroll 8
    for (int j = 0; j < 64; ++j) {
        float nv = n0[(size_t)n * 64 + j];
        sf += nv * wl2s[j * 64 + u];
        sg += nv * wl2s[j * 64 + 32 + u];
        float wv = wl2v[j * 32 + u];
        const float* np = n1 + (size_t)n * 192 + j * 3;
        v0 += np[0] * wv; v1 += np[1] * wv; v2 += np[2] * wv;
    }
    sf *= c1; sg *= c1; v0 *= c1; v1 *= c1; v2 *= c1;

    float rsf = 0.0f, rsg = 0.0f, rv0 = 0.0f, rv1 = 0.0f, rv2 = 0.0f;
#pragma unroll 8
    for (int v = 0; v < MULC; ++v) {
        float h = hs_in[n * MULC + v];
        const float* wp = wrs + (size_t)v * 5 * 64 + sp * 64;
        rsf += h * wp[u];
        rsg += h * wp[32 + u];
        float wv = wrv[(size_t)v * 5 * 32 + sp * 32 + u];
        const float* hvp = hv_in + ((size_t)n * MULC + v) * 3;
        rv0 += hvp[0] * wv; rv1 += hvp[1] * wv; rv2 += hvp[2] * wv;
    }
    sf += rsf * c2; sg += rsg * c2;
    v0 += rv0 * c2; v1 += rv1 * c2; v2 += rv2 * c2;

    float gate = silu(sg);
    hs_out[n * MULC + u] = silu(sf);
    float* ho = hv_out + (size_t)idx * 3;
    ho[0] = v0 * gate; ho[1] = v1 * gate; ho[2] = v2 * gate;
}

// ---------------------------------------------------------------------------
// Readout: [N,32] -> [N,16] -> [N]
// ---------------------------------------------------------------------------
__global__ void k_readout(const float* __restrict__ hs, const float* __restrict__ w1,
                          const float* __restrict__ w2, float* __restrict__ out, int Nn) {
    int n = blockIdx.x * blockDim.x + threadIdx.x;
    if (n >= Nn) return;
    float e = 0.0f;
#pragma unroll
    for (int k = 0; k < 16; ++k) {
        float z = 0.0f;
#pragma unroll 8
        for (int u = 0; u < 32; ++u) z += hs[n * 32 + u] * w1[u * 16 + k];
        e += z * 0.17677669529663687f * w2[k]; // /sqrt(32)
    }
    out[n] = e * 0.25f; // /sqrt(16)
}

// ---------------------------------------------------------------------------
// Host launcher
// ---------------------------------------------------------------------------
extern "C" void kernel_launch(void* const* d_in, const int* in_sizes, int n_in,
                              void* d_out, int out_size, void* d_ws, size_t ws_size,
                              hipStream_t stream) {
    const float* edge_vectors = (const float*)d_in[0];
    const int*   species      = (const int*)d_in[1];
    const int*   senders      = (const int*)d_in[2];
    const int*   receivers    = (const int*)d_in[3];
    const float* embed_w      = (const float*)d_in[4];
    const float* w_res_s      = (const float*)d_in[5];
    const float* w_res_v      = (const float*)d_in[6];
    const float* w_lin1_s     = (const float*)d_in[7];
    const float* w_lin1_v     = (const float*)d_in[8];
    const float* mlp_w0       = (const float*)d_in[9];
    const float* mlp_w1       = (const float*)d_in[10];
    const float* mlp_w2       = (const float*)d_in[11];
    const float* w_lin2_s     = (const float*)d_in[12];
    const float* w_lin2_v     = (const float*)d_in[13];
    const float* w_out1       = (const float*)d_in[14];
    const float* w_out2       = (const float*)d_in[15];
    float* out = (float*)d_out;

    int Nn = in_sizes[1];          // node count
    int Ee = in_sizes[2];          // edge count (multiple of 64 for this problem)

    // Workspace layout
    char*  ws     = (char*)d_ws;
    bf16*  packed = (bf16*)ws;                         // 56 KB of B-fragments
    float* fbase  = (float*)(ws + 65536);
    float* hA_s = fbase;
    float* hA_v = hA_s + (size_t)Nn * 32;
    float* hB_s = hA_v + (size_t)Nn * 96;
    float* hB_v = hB_s + (size_t)Nn * 32;
    float* hs_g = hB_v + (size_t)Nn * 96;
    float* hv_g = hs_g + (size_t)Nn * 32;
    float* n0   = hv_g + (size_t)Nn * 96;
    float* n1   = n0   + (size_t)Nn * 64;

    k_pack<<<2 * FRAGS_PER_LAYER, 32, 0, stream>>>(mlp_w0, mlp_w1, mlp_w2, packed);

    int nb = (Nn * 32 + 127) / 128;
    k_embed<<<nb, 128, 0, stream>>>(species, embed_w, hA_s, hA_v, Nn);

    int edge_blocks = Ee / (16 * WPB);
    const float* hin_s = hA_s; const float* hin_v = hA_v;
    for (int l = 0; l < 2; ++l) {
        float* hout_s = (l == 0) ? hB_s : hA_s;
        float* hout_v = (l == 0) ? hB_v : hA_v;
        k_node_pre<<<nb, 128, 0, stream>>>(hin_s, hin_v,
                                           w_lin1_s + (size_t)l * 32 * 32,
                                           w_lin1_v + (size_t)l * 32 * 32,
                                           hs_g, hv_g, n0, n1, Nn);
        k_edge<<<edge_blocks, 128, 0, stream>>>(edge_vectors, senders, receivers,
                                                packed + (size_t)l * FRAGS_PER_LAYER * FRAG_HALVES,
                                                hs_g, hv_g, n0, n1);
        k_node_post<<<nb, 128, 0, stream>>>(hin_s, hin_v, n0, n1, species,
                                            w_lin2_s + (size_t)l * 64 * 64,
                                            w_lin2_v + (size_t)l * 64 * 32,
                                            w_res_s + (size_t)l * 32 * 5 * 64,
                                            w_res_v + (size_t)l * 32 * 5 * 32,
                                            hout_s, hout_v, Nn);
        hin_s = hout_s; hin_v = hout_v;
    }
    k_readout<<<(Nn + 127) / 128, 128, 0, stream>>>(hin_s, w_out1, w_out2, out, Nn);
}
